// TransAttention2DLayer_15968688406879
// MI455X (gfx1250) — matrix operand order repaired and checked by reading, not compiled, not run on hardware
//
#include <hip/hip_runtime.h>
#include <hip/hip_bf16.h>

// ---------------------------------------------------------------------------
// TransAttention2D for MI455X (gfx1250): bf16 WMMA GEMMs + f32 attention core.
//   B=4, C=256, H=W=64, K=3, EPS=1e-5, scale = C^-0.5 = 0.0625
// ---------------------------------------------------------------------------

#define BB 4
#define CC 256
#define HH 64
#define WW 64
#define NPOS (BB * HH * WW)        // 16384 pixel positions
#define MTILES (NPOS / 16)         // 1024
#define KTILES (CC / 32)           // 8
#define NTILES (CC / 16)           // 16

typedef __attribute__((ext_vector_type(16))) __bf16 v16bf;
typedef __attribute__((ext_vector_type(8)))  float  v8f;

static __device__ __forceinline__ unsigned short f32_to_bf16_rne(float f) {
    unsigned int u = __float_as_uint(f);
    u += 0x7FFFu + ((u >> 16) & 1u);      // round-to-nearest-even
    return (unsigned short)(u >> 16);
}

static __device__ __forceinline__ unsigned int pack_bf16x2(float a, float b) {
    return (unsigned int)f32_to_bf16_rne(a) | ((unsigned int)f32_to_bf16_rne(b) << 16);
}

// A-fragment K index for 16-bit 16x32 A tile (ISA 7.12.2): element e, lane l.
static __device__ __forceinline__ int afrag_k(int e, int l) {
    return (e & 7) + 8 * (((e >> 3) << 1) + (l >> 4));
}

// ---------------------------------------------------------------------------
// 1) LayerNorm stats per (b,c) plane: mu, rsqrt(var+eps)
// ---------------------------------------------------------------------------
__global__ __launch_bounds__(256) void k_ln_stats(const float* __restrict__ x,
                                                  float* __restrict__ mu,
                                                  float* __restrict__ rs) {
    int bc = blockIdx.x;
    const float* p = x + (size_t)bc * (HH * WW);
    float s = 0.f, s2 = 0.f;
    for (int i = threadIdx.x; i < HH * WW; i += 256) {
        float v = p[i];
        s += v; s2 += v * v;
    }
    #pragma unroll
    for (int m = 16; m >= 1; m >>= 1) {
        s  += __shfl_xor(s,  m, 32);
        s2 += __shfl_xor(s2, m, 32);
    }
    __shared__ float as[8], as2[8];
    int wv = threadIdx.x >> 5, ln = threadIdx.x & 31;
    if (ln == 0) { as[wv] = s; as2[wv] = s2; }
    __syncthreads();
    if (threadIdx.x == 0) {
        float S = 0.f, S2 = 0.f;
        #pragma unroll
        for (int i = 0; i < 8; i++) { S += as[i]; S2 += as2[i]; }
        float m_  = S * (1.f / (HH * WW));
        float var = S2 * (1.f / (HH * WW)) - m_ * m_;
        mu[bc] = m_;
        rs[bc] = rsqrtf(var + 1e-5f);
    }
}

// ---------------------------------------------------------------------------
// 2) Normalize + pack xn into bf16 A-fragments [mt][kt][lane][16e]
// ---------------------------------------------------------------------------
__global__ __launch_bounds__(256) void k_norm_pack(const float* __restrict__ x,
                                                   const float* __restrict__ mu,
                                                   const float* __restrict__ rs,
                                                   unsigned short* __restrict__ xnf) {
    int mt = blockIdx.x;                     // m-tile
    int kt = threadIdx.x >> 5;               // k-tile 0..7
    int l  = threadIdx.x & 31;               // frag lane
    int pos = mt * 16 + (l & 15);
    int b = pos >> 12, hw = pos & 4095;

    unsigned int wds[8];
    #pragma unroll
    for (int pr = 0; pr < 8; pr++) {
        int e0 = 2 * pr, e1 = e0 + 1;
        int c0 = kt * 32 + afrag_k(e0, l);
        int c1 = kt * 32 + afrag_k(e1, l);
        int bc0 = b * CC + c0, bc1 = b * CC + c1;
        float v0 = (x[(size_t)bc0 * (HH * WW) + hw] - mu[bc0]) * rs[bc0];
        float v1 = (x[(size_t)bc1 * (HH * WW) + hw] - mu[bc1]) * rs[bc1];
        wds[pr] = pack_bf16x2(v0, v1);
    }
    uint4* d = (uint4*)(xnf + (((size_t)(mt * KTILES + kt)) * 32 + l) * 16);
    d[0] = make_uint4(wds[0], wds[1], wds[2], wds[3]);
    d[1] = make_uint4(wds[4], wds[5], wds[6], wds[7]);
}

// ---------------------------------------------------------------------------
// 3) Pack one 256x256 weight matrix (row-major [out][in]) to B-fragments
//    [nt][kt][lane][16e], B(k,n)=W[n][k].
// ---------------------------------------------------------------------------
__global__ __launch_bounds__(256) void k_pack_w(const float* __restrict__ W,
                                                unsigned short* __restrict__ wf) {
    int t = blockIdx.x * 256 + threadIdx.x;  // (nt*8+kt)*32 + l, 4096 total
    int l = t & 31;
    int ktnt = t >> 5;
    int kt = ktnt & 7, nt = ktnt >> 3;
    int n  = nt * 16 + (l & 15);
    int k0 = kt * 32 + (l >> 4) * 8;         // B-frag: K = (l>>4)*8 + e (e<8), 16 + (l>>4)*8 + (e-8) (e>=8)
    const float* src = W + (size_t)n * CC + k0;
    unsigned int wds[8];
    #pragma unroll
    for (int pr = 0; pr < 4; pr++)
        wds[pr] = pack_bf16x2(src[2 * pr], src[2 * pr + 1]);
    #pragma unroll
    for (int pr = 4; pr < 8; pr++)
        wds[pr] = pack_bf16x2(src[2 * pr + 8], src[2 * pr + 9]);
    uint4* d = (uint4*)(wf + (size_t)t * 16);
    d[0] = make_uint4(wds[0], wds[1], wds[2], wds[3]);
    d[1] = make_uint4(wds[4], wds[5], wds[6], wds[7]);
}

// ---------------------------------------------------------------------------
// 4) Fused Q/K/V projection GEMM. Block = one m-tile, 8 waves; each wave does
//    2 n-tiles x 3 projections = 6 WMMA accumulator tiles.
// ---------------------------------------------------------------------------
__global__ __launch_bounds__(256) void k_qkv_gemm(const unsigned short* __restrict__ xnf,
                                                  const unsigned short* __restrict__ wqf,
                                                  const unsigned short* __restrict__ wkf,
                                                  const unsigned short* __restrict__ wvf,
                                                  const float* __restrict__ bq,
                                                  const float* __restrict__ bk,
                                                  const float* __restrict__ bv,
                                                  float* __restrict__ Qb,
                                                  float* __restrict__ Kb,
                                                  float* __restrict__ Vb) {
    int mt = blockIdx.x;
    int wv = threadIdx.x >> 5;
    int l  = threadIdx.x & 31;
    const unsigned short* wf[3] = { wqf, wkf, wvf };

    v8f acc[6] = {};
    for (int kt = 0; kt < KTILES; kt++) {
        v16bf a = *(const v16bf*)(xnf + (((size_t)(mt * KTILES + kt)) * 32 + l) * 16);
        #pragma unroll
        for (int t = 0; t < 6; t++) {
            int proj = t >> 1;
            int nt = (wv << 1) | (t & 1);
            v16bf bb = *(const v16bf*)(wf[proj] + (((size_t)(nt * KTILES + kt)) * 32 + l) * 16);
            acc[t] = __builtin_amdgcn_wmma_f32_16x16x32_bf16(
                false, a, false, bb, (short)0, acc[t], false, false);
        }
    }

    int Mo = 8 * (l >> 4);
    int N  = l & 15;
    #pragma unroll
    for (int t = 0; t < 6; t++) {
        int proj = t >> 1;
        int nt = (wv << 1) | (t & 1);
        int col = nt * 16 + N;
        float* outb = (proj == 0) ? Qb : (proj == 1) ? Kb : Vb;
        const float* bias = (proj == 0) ? bq : (proj == 1) ? bk : bv;
        float scale = (proj == 0) ? 0.0625f : 1.0f;   // C^-0.5 applied after bias
        float bval = bias[col];
        #pragma unroll
        for (int r = 0; r < 8; r++) {
            int row = mt * 16 + Mo + r;
            outb[(size_t)row * CC + col] = (acc[t][r] + bval) * scale;
        }
    }
}

// ---------------------------------------------------------------------------
// 5) Per-pixel 3x3 neighborhood attention. Wave per pixel, lane owns 8 chans.
//    OOB neighbors contribute K=bk, V=bv (zero-padded input through Linear).
// ---------------------------------------------------------------------------
static __device__ __forceinline__ void load8f(const float* p, float* r) {
    float4 a = ((const float4*)p)[0];
    float4 b = ((const float4*)p)[1];
    r[0] = a.x; r[1] = a.y; r[2] = a.z; r[3] = a.w;
    r[4] = b.x; r[5] = b.y; r[6] = b.z; r[7] = b.w;
}

__global__ __launch_bounds__(256) void k_attn(const float* __restrict__ Qb,
                                              const float* __restrict__ Kb,
                                              const float* __restrict__ Vb,
                                              const float* __restrict__ bk,
                                              const float* __restrict__ bv,
                                              float* __restrict__ ctx) {
    int p = blockIdx.x * 8 + (threadIdx.x >> 5);
    int l = threadIdx.x & 31;
    int b = p >> 12, hw = p & 4095, h = hw >> 6, w = hw & 63;
    int c0 = l * 8;

    float q[8];
    load8f(Qb + (size_t)p * CC + c0, q);

    float logit[9];
    #pragma unroll
    for (int n = 0; n < 9; n++) {
        int dy = n / 3 - 1, dx = n % 3 - 1;
        int hh = h + dy, ww = w + dx;
        bool in = (hh >= 0) & (hh < HH) & (ww >= 0) & (ww < WW);
        const float* kv = in ? (Kb + ((size_t)(b * 4096 + hh * 64 + ww)) * CC + c0)
                             : (bk + c0);
        float kk[8];
        load8f(kv, kk);
        float s = 0.f;
        #pragma unroll
        for (int i = 0; i < 8; i++) s += q[i] * kk[i];
        #pragma unroll
        for (int m = 16; m >= 1; m >>= 1) s += __shfl_xor(s, m, 32);
        logit[n] = s;
    }

    float mx = logit[0];
    #pragma unroll
    for (int n = 1; n < 9; n++) mx = fmaxf(mx, logit[n]);
    float a[9], ssum = 0.f;
    #pragma unroll
    for (int n = 0; n < 9; n++) { a[n] = __expf(logit[n] - mx); ssum += a[n]; }
    float inv = 1.f / ssum;

    float cx[8] = {};
    #pragma unroll
    for (int n = 0; n < 9; n++) {
        int dy = n / 3 - 1, dx = n % 3 - 1;
        int hh = h + dy, ww = w + dx;
        bool in = (hh >= 0) & (hh < HH) & (ww >= 0) & (ww < WW);
        const float* vv = in ? (Vb + ((size_t)(b * 4096 + hh * 64 + ww)) * CC + c0)
                             : (bv + c0);
        float vl[8];
        load8f(vv, vl);
        float wn = a[n] * inv;
        #pragma unroll
        for (int i = 0; i < 8; i++) cx[i] += wn * vl[i];
    }

    float4* d = (float4*)(ctx + (size_t)p * CC + c0);
    d[0] = make_float4(cx[0], cx[1], cx[2], cx[3]);
    d[1] = make_float4(cx[4], cx[5], cx[6], cx[7]);
}

// ---------------------------------------------------------------------------
// 6) Repack ctx (f32 row-major) into bf16 A-fragments for the output GEMM.
// ---------------------------------------------------------------------------
__global__ __launch_bounds__(256) void k_pack_ctx(const float* __restrict__ ctx,
                                                  unsigned short* __restrict__ cf) {
    int mt = blockIdx.x;
    int kt = threadIdx.x >> 5;
    int l  = threadIdx.x & 31;
    int pos = mt * 16 + (l & 15);
    // c(e) = kt*32 + 8*(l>>4) + e           for e<8
    // c(e) = kt*32 + 8*(l>>4) + e + 8       for e>=8
    const float* src = ctx + (size_t)pos * CC + kt * 32 + 8 * (l >> 4);
    unsigned int wds[8];
    #pragma unroll
    for (int pr = 0; pr < 4; pr++)
        wds[pr] = pack_bf16x2(src[2 * pr], src[2 * pr + 1]);
    #pragma unroll
    for (int pr = 4; pr < 8; pr++)
        wds[pr] = pack_bf16x2(src[2 * pr + 8], src[2 * pr + 9]);
    uint4* d = (uint4*)(cf + (((size_t)(mt * KTILES + kt)) * 32 + l) * 16);
    d[0] = make_uint4(wds[0], wds[1], wds[2], wds[3]);
    d[1] = make_uint4(wds[4], wds[5], wds[6], wds[7]);
}

// ---------------------------------------------------------------------------
// 7) Output projection GEMM, writes NCHW [B, OUT_C, H, W] with bias.
// ---------------------------------------------------------------------------
__global__ __launch_bounds__(256) void k_out_gemm(const unsigned short* __restrict__ cf,
                                                  const unsigned short* __restrict__ wof,
                                                  const float* __restrict__ bo,
                                                  float* __restrict__ out) {
    int mt = blockIdx.x;
    int wv = threadIdx.x >> 5;
    int l  = threadIdx.x & 31;

    v8f acc[2] = {};
    for (int kt = 0; kt < KTILES; kt++) {
        v16bf a = *(const v16bf*)(cf + (((size_t)(mt * KTILES + kt)) * 32 + l) * 16);
        #pragma unroll
        for (int t = 0; t < 2; t++) {
            int nt = (wv << 1) | t;
            v16bf bb = *(const v16bf*)(wof + (((size_t)(nt * KTILES + kt)) * 32 + l) * 16);
            acc[t] = __builtin_amdgcn_wmma_f32_16x16x32_bf16(
                false, a, false, bb, (short)0, acc[t], false, false);
        }
    }

    int Mo = 8 * (l >> 4);
    int N  = l & 15;
    #pragma unroll
    for (int t = 0; t < 2; t++) {
        int nt = (wv << 1) | t;
        int oc = nt * 16 + N;
        float bb = bo[oc];
        #pragma unroll
        for (int r = 0; r < 8; r++) {
            int pos = mt * 16 + Mo + r;
            int b = pos >> 12, hw = pos & 4095;
            out[((size_t)b * CC + oc) * 4096 + hw] = acc[t][r] + bb;
        }
    }
}

// ---------------------------------------------------------------------------
// Host launcher
// ---------------------------------------------------------------------------
extern "C" void kernel_launch(void* const* d_in, const int* in_sizes, int n_in,
                              void* d_out, int out_size, void* d_ws, size_t ws_size,
                              hipStream_t stream) {
    (void)in_sizes; (void)n_in; (void)out_size; (void)ws_size;

    const float* x  = (const float*)d_in[0];
    const float* Wq = (const float*)d_in[1];
    const float* bq = (const float*)d_in[2];
    const float* Wk = (const float*)d_in[3];
    const float* bk = (const float*)d_in[4];
    const float* Wv = (const float*)d_in[5];
    const float* bv = (const float*)d_in[6];
    const float* Wo = (const float*)d_in[7];
    const float* bo = (const float*)d_in[8];
    float* out = (float*)d_out;

    char* ws = (char*)d_ws;
    const size_t SZ_WF  = (size_t)NTILES * KTILES * 32 * 16 * 2;  // 128 KB / matrix
    const size_t SZ_XN  = (size_t)MTILES * KTILES * 32 * 16 * 2;  // 8 MB
    const size_t SZ_ACT = (size_t)NPOS * CC * 4;                  // 16 MB

    float*          mu  = (float*)(ws + 0);
    float*          rs  = (float*)(ws + 4096);
    unsigned short* wqf = (unsigned short*)(ws + 8192);
    unsigned short* wkf = (unsigned short*)(ws + 8192 + SZ_WF);
    unsigned short* wvf = (unsigned short*)(ws + 8192 + 2 * SZ_WF);
    unsigned short* wof = (unsigned short*)(ws + 8192 + 3 * SZ_WF);
    unsigned short* xnf = (unsigned short*)(ws + 8192 + 4 * SZ_WF);
    float*          Qb  = (float*)(ws + 8192 + 4 * SZ_WF + SZ_XN);
    float*          Kb  = (float*)((char*)Qb + SZ_ACT);
    float*          Vb  = (float*)((char*)Kb + SZ_ACT);
    float*          ctx = (float*)((char*)Vb + SZ_ACT);
    unsigned short* cf  = (unsigned short*)((char*)ctx + SZ_ACT);

    k_ln_stats<<<BB * CC, 256, 0, stream>>>(x, mu, rs);
    k_norm_pack<<<MTILES, 256, 0, stream>>>(x, mu, rs, xnf);
    k_pack_w<<<16, 256, 0, stream>>>(Wq, wqf);
    k_pack_w<<<16, 256, 0, stream>>>(Wk, wkf);
    k_pack_w<<<16, 256, 0, stream>>>(Wv, wvf);
    k_pack_w<<<16, 256, 0, stream>>>(Wo, wof);
    k_qkv_gemm<<<MTILES, 256, 0, stream>>>(xnf, wqf, wkf, wvf, bq, bk, bv, Qb, Kb, Vb);
    k_attn<<<NPOS / 8, 256, 0, stream>>>(Qb, Kb, Vb, bk, bv, ctx);
    k_pack_ctx<<<MTILES, 256, 0, stream>>>(ctx, cf);
    k_out_gemm<<<MTILES, 256, 0, stream>>>(cf, wof, bo, out);
}
